// ViT_53730040873225
// MI455X (gfx1250) — matrix operand order repaired and checked by reading, not compiled, not run on hardware
//
#include <hip/hip_runtime.h>
#include <hip/hip_bf16.h>
#include <math.h>

// ---------------------------------------------------------------- types
typedef __attribute__((ext_vector_type(8)))  __bf16 v8bf;
typedef __attribute__((ext_vector_type(16))) __bf16 v16bf;
typedef __attribute__((ext_vector_type(8)))  float  v8f;

constexpr int Bsz = 32;
constexpr int Pn  = 196;
constexpr int PDim = 768;
constexpr int Dm  = 768;
constexpr int Hn  = 12;
constexpr int Ln  = 8;
constexpr int NCn = 1000;
constexpr int HDn = 64;
constexpr int Sn  = 197;                // P + 1 (cls token)
constexpr int BSn = Bsz * Sn;           // 6304 rows (multiple of 16)
constexpr int SP  = 224;                // padded key dim (multiple of 32)

// ---------------------------------------------------------------- helpers
__device__ __forceinline__ __bf16 f2bf(float f) {
  unsigned u = __builtin_bit_cast(unsigned, f);
  unsigned r = u + 0x7FFFu + ((u >> 16) & 1u);   // round-to-nearest-even
  unsigned short hs = (unsigned short)(r >> 16);
  return __builtin_bit_cast(__bf16, hs);
}

__device__ __forceinline__ v8f wmma_bf16(v16bf a, v16bf b, v8f c) {
  return __builtin_amdgcn_wmma_f32_16x16x32_bf16(
      false, a, false, b, (short)0, c, false, false);
}

// A fragment (16x32 bf16) from row-major bf16 [M][lda].
// ISA 7.12.2: lane<16 -> M=lane, halves 0..7 = K0..7, halves 8..15 = K16..23;
// lanes 16-31 same M, K offset +8.  Two contiguous 16B loads, branch-free clamp.
__device__ __forceinline__ v16bf ldA16(const __bf16* __restrict__ A, int lda,
                                       int r0, int c0, int rmax) {
  int lane = threadIdx.x & 31;
  int r = r0 + (lane & 15);
  r = r < rmax ? r : rmax - 1;
  const __bf16* p = A + (long)r * lda + c0 + ((lane >> 4) << 3);
  v8bf lo = *(const v8bf*)p;          // K +0..7
  v8bf hi = *(const v8bf*)(p + 16);   // K +16..23
  return __builtin_shufflevector(lo, hi, 0, 1, 2, 3, 4, 5, 6, 7,
                                 8, 9, 10, 11, 12, 13, 14, 15);
}

// B fragment (32x16 bf16) from TRANSPOSED bf16 [N][ld] (row n = output col).
// lane<16 -> N=lane, elements 0..7 = K0..7, 8..15 = K16..23; lanes 16-31 K +8.
// Two contiguous 16B loads per lane.
__device__ __forceinline__ v16bf ldB16(const __bf16* __restrict__ Mt, int ld,
                                       int k0, int n0, int nmax) {
  int lane = threadIdx.x & 31;
  int n = n0 + (lane & 15);
  n = n < nmax ? n : nmax - 1;
  const __bf16* p = Mt + (long)n * ld + k0 + ((lane >> 4) << 3);
  v8bf lo = *(const v8bf*)p;
  v8bf hi = *(const v8bf*)(p + 16);
  return __builtin_shufflevector(lo, hi, 0, 1, 2, 3, 4, 5, 6, 7,
                                 8, 9, 10, 11, 12, 13, 14, 15);
}

// A fragment from LDS bf16 probability tile (16 x SP) -> ds_load_b128 x2.
__device__ __forceinline__ v16bf ldA_lds(const __bf16 (*pb)[SP], int k0) {
  int lane = threadIdx.x & 31;
  const __bf16* p = &pb[lane & 15][k0 + ((lane >> 4) << 3)];
  v8bf lo = *(const v8bf*)p;
  v8bf hi = *(const v8bf*)(p + 16);
  return __builtin_shufflevector(lo, hi, 0, 1, 2, 3, 4, 5, 6, 7,
                                 8, 9, 10, 11, 12, 13, 14, 15);
}

// ---------------------------------------------------------------- setup
__global__ void cvt_kernel(const float* __restrict__ X,
                           __bf16* __restrict__ Y, long n) {
  long i = (long)blockIdx.x * 256 + threadIdx.x;
  if (i < n) Y[i] = f2bf(X[i]);
}

// batched transpose+convert: W [batch][K][N] f32 -> WT [batch][N][K] bf16
__global__ void transposeT_kernel(const float* __restrict__ W,
                                  __bf16* __restrict__ WT, int K, int N,
                                  long total) {
  long idx = (long)blockIdx.x * 256 + threadIdx.x;
  if (idx >= total) return;
  long bkn = (long)K * N;
  long b = idx / bkn;
  long rem = idx - b * bkn;
  int k = (int)(rem / N), n = (int)(rem - (long)k * N);
  WT[b * bkn + (long)n * K + k] = f2bf(W[idx]);
}

__global__ void zero_kernel(float* __restrict__ p, long n) {
  long i = (long)blockIdx.x * 256 + threadIdx.x;
  if (i < n) p[i] = 0.f;
}

__global__ void clstok_kernel(const float* __restrict__ vcls,
                              float* __restrict__ h) {
  int i = blockIdx.x * 256 + threadIdx.x;
  if (i >= Bsz * Dm) return;
  int b = i / Dm, d = i - b * Dm;
  h[(long)b * Sn * Dm + d] = vcls[d];
}

__global__ void clsext_kernel(const float* __restrict__ h,
                              __bf16* __restrict__ cls) {
  int i = blockIdx.x * 256 + threadIdx.x;
  if (i >= Bsz * Dm) return;
  int b = i / Dm, d = i - b * Dm;
  cls[i] = f2bf(h[(long)b * Sn * Dm + d]);
}

// ---------------------------------------------------------------- GEMM
// 64x64 register tile per wave: 4 A-frags x 4 B-frags -> 16 WMMA per K-step
// (32 FLOP per operand byte).  MODE is compile-time so each epilogue's
// register demand stays lean (avoids accumulator spill-to-scratch):
// MODE 0: f32 out (+optional residual R)   MODE 1: exact GELU -> bf16 out
// MODE 2: patch-embed (row remap b*197+p+1, + sin/cos positional) -> f32 out
template <int MODE>
__global__ __launch_bounds__(256) void gemm_kernel(
    const __bf16* __restrict__ A, int lda, int acol,
    const __bf16* __restrict__ WT, int ldw,
    const float* __restrict__ bias,
    const float* __restrict__ R, long ldr,
    float* __restrict__ Cf, __bf16* __restrict__ Cb, long ldc, int ccol,
    int M, int N, int K, int ntn) {
  int wid = blockIdx.x * 8 + (threadIdx.x >> 5);
  int mt = wid / ntn, nt = wid - mt * ntn;
  if (mt * 64 >= M) return;            // whole-wave uniform exit: EXEC stays full
  int r0 = mt * 64, n0 = nt * 64;
  v8f acc[4][4] = {};
  for (int k0 = 0; k0 < K; k0 += 32) {
    v16bf a[4], b[4];
#pragma unroll
    for (int i = 0; i < 4; ++i)
      a[i] = ldA16(A, lda, r0 + 16 * i, acol + k0, M);
#pragma unroll
    for (int j = 0; j < 4; ++j)
      b[j] = ldB16(WT, ldw, k0, n0 + 16 * j, N);
#pragma unroll
    for (int i = 0; i < 4; ++i)
#pragma unroll
      for (int j = 0; j < 4; ++j)
        acc[i][j] = wmma_bf16(a[i], b[j], acc[i][j]);
  }
  int lane = threadIdx.x & 31;
#pragma unroll
  for (int j = 0; j < 4; ++j) {
    int n = n0 + 16 * j + (lane & 15);
    if (n >= N) continue;
    float bv = bias ? bias[n] : 0.f;
#pragma unroll
    for (int i = 0; i < 4; ++i) {
#pragma unroll
      for (int r = 0; r < 8; ++r) {
        int m = r0 + 16 * i + r + ((lane >> 4) << 3);
        if (m >= M) continue;
        float val = acc[i][j][r] + bv;
        if (MODE == 2) {                // patch embedding epilogue
          int bb = m / Pn, p = m - bb * Pn;
          long orow = (long)bb * Sn + p + 1;
          float e = (p & 1) ? (float)(n - 1) : (float)n;  // parity on POSITION
          float ang = (float)p * __powf(10000.f, -e / (float)Dm);
          val += (p & 1) ? __cosf(ang) : __sinf(ang);
          Cf[orow * ldc + ccol + n] = val;
        } else if (MODE == 1) {         // exact GELU -> bf16
          val = 0.5f * val * (1.f + erff(val * 0.70710678118f));
          Cb[(long)m * ldc + ccol + n] = f2bf(val);
        } else {
          if (R) val += R[(long)m * ldr + ccol + n];
          Cf[(long)m * ldc + ccol + n] = val;
        }
      }
    }
  }
}

// ---------------------------------------------------------------- QKV
// per-head 64x64 projection as one 64x64 register tile per wave
// (2 K-steps x 16 WMMA).  WHICH is compile-time: 0 -> q, 1 -> k (row-major
// bf16), 2 -> v stored TRANSPOSED as vT[b][d][s] (stride SP) so the P@V
// B-fragments are contiguous.
template <int WHICH>
__global__ __launch_bounds__(256) void qkv_kernel(
    const __bf16* __restrict__ xn, const __bf16* __restrict__ WTbase,
    const float* __restrict__ bbase, __bf16* __restrict__ outp, int layer) {
  int head = blockIdx.y;
  const __bf16* WT = WTbase + (long)(layer * Hn + head) * HDn * HDn;
  const float*  bi = bbase + (long)(layer * Hn + head) * HDn;

  int wid = blockIdx.x * 8 + (threadIdx.x >> 5);
  if (wid * 64 >= BSn) return;
  int r0 = wid * 64, acol = head * HDn;
  v8f acc[4][4] = {};
  for (int k0 = 0; k0 < HDn; k0 += 32) {
    v16bf a[4], b[4];
#pragma unroll
    for (int i = 0; i < 4; ++i)
      a[i] = ldA16(xn, Dm, r0 + 16 * i, acol + k0, BSn);
#pragma unroll
    for (int j = 0; j < 4; ++j)
      b[j] = ldB16(WT, HDn, k0, 16 * j, HDn);
#pragma unroll
    for (int i = 0; i < 4; ++i)
#pragma unroll
      for (int j = 0; j < 4; ++j)
        acc[i][j] = wmma_bf16(a[i], b[j], acc[i][j]);
  }
  int lane = threadIdx.x & 31;
#pragma unroll
  for (int j = 0; j < 4; ++j) {
    int n = 16 * j + (lane & 15);
    float bvv = bi[n];
#pragma unroll
    for (int i = 0; i < 4; ++i) {
#pragma unroll
      for (int r = 0; r < 8; ++r) {
        int m = r0 + 16 * i + r + ((lane >> 4) << 3);
        if (m >= BSn) continue;
        __bf16 val = f2bf(acc[i][j][r] + bvv);
        if (WHICH == 2) {
          int bb = m / Sn, s = m - bb * Sn;
          outp[((long)bb * Dm + acol + n) * SP + s] = val;
        } else {
          outp[(long)m * Dm + acol + n] = val;
        }
      }
    }
  }
}

// ---------------------------------------------------------------- attention
// one wave per (b, head, 16-query tile): WMMA scores -> LDS f32, softmax,
// head-mean into attns, bf16 probs -> LDS, WMMA P@V, h += O.
__global__ __launch_bounds__(32) void attn_kernel(
    const __bf16* __restrict__ q, const __bf16* __restrict__ k,
    const __bf16* __restrict__ vT, float* __restrict__ h,
    float* __restrict__ attn) {
  int qt = blockIdx.x, head = blockIdx.y, b = blockIdx.z;
  int lane = threadIdx.x & 31;
  __shared__ float  sc[16][SP];
  __shared__ __bf16 pb[16][SP];
  int r0 = qt * 16;
  long rowbase = (long)b * Sn;
  const __bf16* qb = q + rowbase * Dm;
  const __bf16* kb = k + rowbase * Dm;
  const __bf16* vb = vT + (long)b * Dm * SP;

  // scores = (Q @ K^T) / 8 over 13 key tiles (row/key clamps are branch-free;
  // clamped-duplicate scores land in masked columns/rows)
  for (int nt = 0; nt < 13; ++nt) {
    v8f acc = {};
    for (int kk = 0; kk < HDn; kk += 32) {
      v16bf a  = ldA16(qb, Dm, r0, head * HDn + kk, Sn);
      v16bf bf = ldB16(kb, Dm, head * HDn + kk, nt * 16, Sn);
      acc = wmma_bf16(a, bf, acc);
    }
    int n = nt * 16 + (lane & 15);
#pragma unroll
    for (int r = 0; r < 8; ++r) {
      int m = r + ((lane >> 4) << 3);
      sc[m][n] = acc[r] * 0.125f;       // 1/sqrt(64)
    }
  }
  __syncthreads();
  for (int c = Sn + lane; c < SP; c += 32)    // mask invalid key columns
    for (int m = 0; m < 16; ++m) sc[m][c] = -1e30f;
  __syncthreads();

  // row softmax; emit bf16 probs to LDS and head-mean to attns
  for (int m = 0; m < 16; ++m) {
    float mx = -1e30f;
    for (int c = lane; c < SP; c += 32) mx = fmaxf(mx, sc[m][c]);
    for (int o = 16; o; o >>= 1) mx = fmaxf(mx, __shfl_xor(mx, o));
    float sum = 0.f;
    for (int c = lane; c < SP; c += 32) {
      float e = __expf(sc[m][c] - mx);
      sc[m][c] = e; sum += e;
    }
    for (int o = 16; o; o >>= 1) sum += __shfl_xor(sum, o);
    float inv = 1.f / sum;
    int row = r0 + m;
    for (int c = lane; c < SP; c += 32) {
      float p = sc[m][c] * inv;
      pb[m][c] = f2bf(p);
      if (row < Sn && c < Sn)
        atomicAdd(attn + ((long)b * Sn + row) * Sn + c, p * (1.f / Hn));
    }
  }
  __syncthreads();

  // O = P @ V  (B-fragment contiguous from vT); residual add into h
  for (int ct = 0; ct < 4; ++ct) {
    v8f acc = {};
    for (int k0 = 0; k0 < SP; k0 += 32) {
      v16bf a  = ldA_lds(pb, k0);
      v16bf bf = ldB16(vb, SP, k0, head * HDn + ct * 16, Dm);
      acc = wmma_bf16(a, bf, acc);
    }
    int col = head * HDn + ct * 16 + (lane & 15);
#pragma unroll
    for (int r = 0; r < 8; ++r) {
      int m = r0 + r + ((lane >> 4) << 3);
      if (m < Sn) h[(rowbase + m) * (long)Dm + col] += acc[r];
    }
  }
}

// ---------------------------------------------------------------- layernorm
__global__ __launch_bounds__(256) void ln_kernel(
    const float* __restrict__ X, const float* __restrict__ g,
    const float* __restrict__ bta, __bf16* __restrict__ Y) {
  long row = blockIdx.x;
  const float* x = X + row * Dm;
  float s = 0.f, s2 = 0.f;
  for (int i = threadIdx.x; i < Dm; i += 256) {
    float u = x[i]; s += u; s2 += u * u;
  }
  for (int o = 16; o; o >>= 1) { s += __shfl_xor(s, o); s2 += __shfl_xor(s2, o); }
  __shared__ float sh[16];
  int w = threadIdx.x >> 5, lane = threadIdx.x & 31;
  if (lane == 0) { sh[w] = s; sh[w + 8] = s2; }
  __syncthreads();
  float ts = 0.f, ts2 = 0.f;
  for (int i = 0; i < 8; ++i) { ts += sh[i]; ts2 += sh[i + 8]; }
  float mean = ts / Dm;
  float var  = ts2 / Dm - mean * mean;
  float inv  = rsqrtf(var + 1e-5f);
  for (int i = threadIdx.x; i < Dm; i += 256)
    Y[row * Dm + i] = f2bf((x[i] - mean) * inv * g[i] + bta[i]);
}

// ---------------------------------------------------------------- softmax
__global__ __launch_bounds__(256) void softmax_kernel(float* __restrict__ logits,
                                                      int n) {
  float* x = logits + (long)blockIdx.x * n;
  __shared__ float sh[8];
  int w = threadIdx.x >> 5, lane = threadIdx.x & 31;
  float mx = -1e30f;
  for (int i = threadIdx.x; i < n; i += 256) mx = fmaxf(mx, x[i]);
  for (int o = 16; o; o >>= 1) mx = fmaxf(mx, __shfl_xor(mx, o));
  if (lane == 0) sh[w] = mx;
  __syncthreads();
  mx = sh[0];
  for (int i = 1; i < 8; ++i) mx = fmaxf(mx, sh[i]);
  __syncthreads();
  float sum = 0.f;
  for (int i = threadIdx.x; i < n; i += 256) {
    float e = __expf(x[i] - mx); x[i] = e; sum += e;
  }
  for (int o = 16; o; o >>= 1) sum += __shfl_xor(sum, o);
  if (lane == 0) sh[w] = sum;
  __syncthreads();
  sum = 0.f;
  for (int i = 0; i < 8; ++i) sum += sh[i];
  float inv = 1.f / sum;
  for (int i = threadIdx.x; i < n; i += 256) x[i] *= inv;
}

// ---------------------------------------------------------------- launcher
extern "C" void kernel_launch(void* const* d_in, const int* in_sizes, int n_in,
                              void* d_out, int out_size, void* d_ws,
                              size_t ws_size, hipStream_t stream) {
  (void)in_sizes; (void)n_in; (void)out_size; (void)ws_size;
  const float* x      = (const float*)d_in[0];
  const float* W_emb  = (const float*)d_in[1];
  const float* b_emb  = (const float*)d_in[2];
  const float* v_cls  = (const float*)d_in[3];
  const float* ln1_g  = (const float*)d_in[4];
  const float* ln1_b  = (const float*)d_in[5];
  const float* Wq     = (const float*)d_in[6];
  const float* bq     = (const float*)d_in[7];
  const float* Wk     = (const float*)d_in[8];
  const float* bk     = (const float*)d_in[9];
  const float* Wv     = (const float*)d_in[10];
  const float* bv     = (const float*)d_in[11];
  const float* ln2_g  = (const float*)d_in[12];
  const float* ln2_b  = (const float*)d_in[13];
  const float* W1     = (const float*)d_in[14];
  const float* b1     = (const float*)d_in[15];
  const float* W2     = (const float*)d_in[16];
  const float* b2     = (const float*)d_in[17];
  const float* W_cls  = (const float*)d_in[18];
  const float* b_cls  = (const float*)d_in[19];

  float* out   = (float*)d_out;
  float* pred  = out;                       // [32,1000]
  float* attns = out + (long)Bsz * NCn;     // [L,B,S,S]

  // ---- workspace carve-up (256B aligned)
  char* base = (char*)d_ws;
  size_t off = 0;
  auto alloc = [&](size_t bytes) -> void* {
    off = (off + 255) & ~(size_t)255;
    void* p = base + off;
    off += bytes;
    return p;
  };
  float*  h     = (float*) alloc((size_t)BSn * Dm * 4);
  __bf16* xn    = (__bf16*)alloc((size_t)BSn * Dm * 2);
  __bf16* qb    = (__bf16*)alloc((size_t)BSn * Dm * 2);
  __bf16* kb    = (__bf16*)alloc((size_t)BSn * Dm * 2);
  __bf16* vT    = (__bf16*)alloc((size_t)Bsz * Dm * SP * 2);
  __bf16* mh    = (__bf16*)alloc((size_t)BSn * 4 * Dm * 2);
  __bf16* xbf   = (__bf16*)alloc((size_t)Bsz * Pn * PDim * 2);
  __bf16* clsbf = (__bf16*)alloc((size_t)Bsz * Dm * 2);
  __bf16* WembT = (__bf16*)alloc((size_t)Dm * PDim * 2);
  __bf16* WqT   = (__bf16*)alloc((size_t)Ln * Hn * HDn * HDn * 2);
  __bf16* WkT   = (__bf16*)alloc((size_t)Ln * Hn * HDn * HDn * 2);
  __bf16* WvT   = (__bf16*)alloc((size_t)Ln * Hn * HDn * HDn * 2);
  __bf16* W1T   = (__bf16*)alloc((size_t)Ln * 4 * Dm * Dm * 2);
  __bf16* W2T   = (__bf16*)alloc((size_t)Ln * Dm * 4 * Dm * 2);
  __bf16* WclsT = (__bf16*)alloc((size_t)NCn * Dm * 2);

  auto grid1 = [](long n) { return (int)((n + 255) / 256); };

  // ---- one-time (per launch) weight prep: transpose + bf16 convert
  long nx = (long)Bsz * Pn * PDim;
  cvt_kernel<<<grid1(nx), 256, 0, stream>>>(x, xbf, nx);
  long nWemb = (long)PDim * Dm;
  transposeT_kernel<<<grid1(nWemb), 256, 0, stream>>>(W_emb, WembT, PDim, Dm, nWemb);
  long nWh = (long)Ln * Hn * HDn * HDn;
  transposeT_kernel<<<grid1(nWh), 256, 0, stream>>>(Wq, WqT, HDn, HDn, nWh);
  transposeT_kernel<<<grid1(nWh), 256, 0, stream>>>(Wk, WkT, HDn, HDn, nWh);
  transposeT_kernel<<<grid1(nWh), 256, 0, stream>>>(Wv, WvT, HDn, HDn, nWh);
  long nW1 = (long)Ln * Dm * 4 * Dm;
  transposeT_kernel<<<grid1(nW1), 256, 0, stream>>>(W1, W1T, Dm, 4 * Dm, nW1);
  transposeT_kernel<<<grid1(nW1), 256, 0, stream>>>(W2, W2T, 4 * Dm, Dm, nW1);
  long nWc = (long)Dm * NCn;
  transposeT_kernel<<<grid1(nWc), 256, 0, stream>>>(W_cls, WclsT, Dm, NCn, nWc);

  auto gemm = [&](const __bf16* A, int lda, int acol, const __bf16* WT, int ldw,
                  const float* bias, const float* R, long ldr, float* Cf,
                  __bf16* Cb, long ldc, int ccol, int M, int N, int K,
                  int mode) {
    int ntn = (N + 63) / 64;
    int ntm = (M + 63) / 64;
    long tiles = (long)ntm * ntn;
    int blocks = (int)((tiles + 7) / 8);
    if (mode == 0)
      gemm_kernel<0><<<blocks, 256, 0, stream>>>(A, lda, acol, WT, ldw, bias, R,
                                                 ldr, Cf, Cb, ldc, ccol, M, N,
                                                 K, ntn);
    else if (mode == 1)
      gemm_kernel<1><<<blocks, 256, 0, stream>>>(A, lda, acol, WT, ldw, bias, R,
                                                 ldr, Cf, Cb, ldc, ccol, M, N,
                                                 K, ntn);
    else
      gemm_kernel<2><<<blocks, 256, 0, stream>>>(A, lda, acol, WT, ldw, bias, R,
                                                 ldr, Cf, Cb, ldc, ccol, M, N,
                                                 K, ntn);
  };

  // ---- patch + positional embedding (mode 2 remaps rows to b*197+p+1)
  gemm(xbf, PDim, 0, WembT, PDim, b_emb, nullptr, 0, h, nullptr, Dm, 0,
       Bsz * Pn, Dm, PDim, 2);
  clstok_kernel<<<grid1(Bsz * Dm), 256, 0, stream>>>(v_cls, h);

  long nattn = (long)Ln * Bsz * Sn * Sn;
  zero_kernel<<<grid1(nattn), 256, 0, stream>>>(attns, nattn);

  // ---- transformer layers
  for (int l = 0; l < Ln; ++l) {
    ln_kernel<<<BSn, 256, 0, stream>>>(h, ln1_g + (long)l * Dm,
                                       ln1_b + (long)l * Dm, xn);
    dim3 qg(13, Hn);                     // ceil(99 64-row tiles / 8 waves)
    qkv_kernel<0><<<qg, 256, 0, stream>>>(xn, WqT, bq, qb, l);
    qkv_kernel<1><<<qg, 256, 0, stream>>>(xn, WkT, bk, kb, l);
    qkv_kernel<2><<<qg, 256, 0, stream>>>(xn, WvT, bv, vT, l);
    dim3 ag(13, Hn, Bsz);
    attn_kernel<<<ag, 32, 0, stream>>>(qb, kb, vT, h,
                                       attns + (long)l * Bsz * Sn * Sn);
    ln_kernel<<<BSn, 256, 0, stream>>>(h, ln2_g + (long)l * Dm,
                                       ln2_b + (long)l * Dm, xn);
    gemm(xn, Dm, 0, W1T + (long)l * 4 * Dm * Dm, Dm, b1 + (long)l * 4 * Dm,
         nullptr, 0, nullptr, mh, 4 * Dm, 0, BSn, 4 * Dm, Dm, 1);  // GELU->bf16
    gemm(mh, 4 * Dm, 0, W2T + (long)l * Dm * 4 * Dm, 4 * Dm,
         b2 + (long)l * Dm, h, Dm, h, nullptr, Dm, 0, BSn, Dm, 4 * Dm, 0);
  }

  // ---- classifier on cls rows
  clsext_kernel<<<grid1(Bsz * Dm), 256, 0, stream>>>(h, clsbf);
  gemm(clsbf, Dm, 0, WclsT, Dm, b_cls, nullptr, 0, pred, nullptr, NCn, 0, Bsz,
       NCn, Dm, 0);
  softmax_kernel<<<Bsz, 256, 0, stream>>>(pred, NCn);
}
